// KNet_55259049230647
// MI455X (gfx1250) — compile-verified, matrix operand order checked by reading, not run on hardware
//
#include <hip/hip_runtime.h>
#include <stdint.h>

typedef __attribute__((ext_vector_type(16))) __bf16 v16bf;
typedef __attribute__((ext_vector_type(8)))  float  v8f;

// GCC-style int4 vector to match the async-LDS builtin's parameter type
typedef int v4i_ __attribute__((vector_size(4 * sizeof(int))));
typedef __attribute__((address_space(1))) v4i_* gv4i_p;
typedef __attribute__((address_space(3))) v4i_* lv4i_p;

#define BN_EPS 1e-5f

// Problem geometry (fixed by the reference)
#define Bn   2
#define CIN0 6
#define Cch  64
#define Hh   512
#define Ww   512
#define Rres 4
#define TILE 64            // pixels per sub-tile along W
#define TPW  (Ww / TILE)   // 8 sub-tiles per row
#define ACTE (3 * 66 * Cch) // act halo buffer elements (bf16)

// ------------------------------------------------------------------
// helpers
// ------------------------------------------------------------------
__device__ __forceinline__ void wait_async_lds() {
#if __has_builtin(__builtin_amdgcn_s_wait_asynccnt)
  __builtin_amdgcn_s_wait_asynccnt(0);
#else
  asm volatile("s_wait_asynccnt 0" ::: "memory");
#endif
}

__device__ __forceinline__ void lds_fill16(__bf16* dst, const __bf16* gsrc) {
#if __has_builtin(__builtin_amdgcn_global_load_async_to_lds_b128)
  __builtin_amdgcn_global_load_async_to_lds_b128(
      (gv4i_p)(uintptr_t)gsrc, (lv4i_p)(uintptr_t)dst, 0, 0);
#else
  *(uint4*)dst = *(const uint4*)gsrc;
#endif
}

// stage one 3-row x 66-px x 64-ch bf16 halo tile into an LDS buffer
__device__ __forceinline__ void stage_act(__bf16* dstbuf, const __bf16* in,
                                          int rowid, int h, int x0, int tid) {
  const int NV = 3 * 66 * 8;                 // 16-byte chunks
  for (int i = tid; i < NV; i += 128) {
    int c8 = i & 7;
    int xi = (i >> 3) % 66;
    int r  = i / (8 * 66);
    __bf16* dst = &dstbuf[(size_t)((r * 66 + xi) * Cch) + c8 * 8];
    int hs = h + r - 1;
    int xg = x0 + xi - 1;
    if (hs >= 0 && hs < Hh && xg >= 0 && xg < Ww) {
      lds_fill16(dst, in + ((size_t)(rowid + (r - 1)) * Ww + xg) * Cch + c8 * 8);
    } else {
      *(uint4*)dst = make_uint4(0u, 0u, 0u, 0u);
    }
  }
}

// ------------------------------------------------------------------
// prep: fold BN into scale/bias
// ------------------------------------------------------------------
__global__ void bn_prep_kernel(const float* g1, const float* b1, const float* m1, const float* v1,
                               const float* g2, const float* b2, const float* m2, const float* v2,
                               float* s1, float* o1, float* s2, float* o2, int n) {
  int i = blockIdx.x * blockDim.x + threadIdx.x;
  if (i >= n) return;
  float s = g1[i] * rsqrtf(v1[i] + BN_EPS);
  s1[i] = s; o1[i] = b1[i] - m1[i] * s;
  s = g2[i] * rsqrtf(v2[i] + BN_EPS);
  s2[i] = s; o2[i] = b2[i] - m2[i] * s;
}

// ------------------------------------------------------------------
// prep: repack f32 OIHW (coutr x 64 x 3 x 3) into bf16 WMMA B-fragments.
// Layout: [tap(9)][kc(2)][nt(NT)][lane(32)][elem(16)], per ISA 16-bit
// 32x16 B layout: lanes 0-15 -> N, VGPR0-3 K0..7 (grp0)/K8..15 (grp1),
// VGPR4-7 K16..23 (grp0)/K24..31 (grp1).
// ------------------------------------------------------------------
__global__ void pack_w3_kernel(const float* __restrict__ src, __bf16* __restrict__ dst,
                               int NT, int coutr) {
  int idx = blockIdx.x * blockDim.x + threadIdx.x;
  int total = 9 * 2 * NT * 32 * 16;
  if (idx >= total) return;
  int i    = idx & 15;
  int lane = (idx >> 4) & 31;
  int nt   = (idx >> 9) % NT;
  int rest = idx / (512 * NT);
  int kc   = rest & 1;
  int tap  = rest >> 1;
  int dy = tap / 3, dx = tap % 3;
  int n   = nt * 16 + (lane & 15);
  int grp = lane >> 4;
  int v = i >> 1, h = i & 1;
  int k = ((v < 4) ? 0 : 16) + (grp ? 8 : 0) + (v & 3) * 2 + h;
  int cin = kc * 32 + k;
  float val = 0.f;
  if (n < coutr) val = src[((size_t)(n * Cch + cin) * 3 + dy) * 3 + dx];
  dst[idx] = (__bf16)val;
}

// ------------------------------------------------------------------
// conv_first: 7x7, 6 -> 64, ReLU, f32 NCHW in -> bf16 NHWC out
// block = 256 threads = 32 pixels x 8 cout-groups of 8
// ------------------------------------------------------------------
__global__ void conv_first_kernel(const float* __restrict__ x, const float* __restrict__ w,
                                  __bf16* __restrict__ out) {
  __shared__ __bf16 wl[Cch * CIN0 * 49];     // 37632 B
  int tid = threadIdx.x;
  for (int i = tid; i < Cch * CIN0 * 49; i += 256) wl[i] = (__bf16)w[i];
  __syncthreads();

  int p     = blockIdx.x * 32 + (tid & 31);
  int cog   = tid >> 5;                      // 0..7
  int xw    = p % Ww;
  int rowid = p / Ww;                        // b*H + h
  int b = rowid / Hh, h = rowid % Hh;

  float acc[8];
#pragma unroll
  for (int o = 0; o < 8; ++o) acc[o] = 0.f;

  for (int cin = 0; cin < CIN0; ++cin) {
    const float* xc = x + (size_t)(b * CIN0 + cin) * Hh * Ww;
    for (int dy = 0; dy < 7; ++dy) {
      int ys = h + dy - 3;
      if (ys < 0 || ys >= Hh) continue;
      for (int dx = 0; dx < 7; ++dx) {
        int xs = xw + dx - 3;
        if (xs < 0 || xs >= Ww) continue;
        float xv = xc[(size_t)ys * Ww + xs];
        const __bf16* wp = &wl[((cog * 8) * CIN0 + cin) * 49 + dy * 7 + dx];
#pragma unroll
        for (int o = 0; o < 8; ++o)
          acc[o] += xv * (float)wp[(size_t)o * CIN0 * 49];
      }
    }
  }
  size_t ob = ((size_t)rowid * Ww + xw) * Cch + cog * 8;
#pragma unroll
  for (int o = 0; o < 8; ++o)
    out[ob + o] = (__bf16)fmaxf(acc[o], 0.f);
}

// ------------------------------------------------------------------
// Main 3x3 implicit-GEMM WMMA conv, Cin=64 bf16 NHWC.
// One block = one image row (512 px), 128 threads (4 waves).
// LDS: packed weights (loaded once) + double-buffered act halo tiles,
// filled by async-to-LDS so the next tile's DMA overlaps WMMA compute.
// Inner loop batches A + all NT B fragment loads before the WMMA burst
// so the DS waits amortize over NT matrix ops.
// ------------------------------------------------------------------
template <int NT, bool OUT_F32>
__global__ void conv3x3_wmma_kernel(const __bf16* __restrict__ in,    // NHWC C=64
                                    const __bf16* __restrict__ wp,    // packed B fragments
                                    const float*  __restrict__ scale, // per-cout (or null)
                                    const float*  __restrict__ bias,
                                    const __bf16* __restrict__ resid, // NHWC C=64 (or null)
                                    __bf16* __restrict__ out_bf,
                                    float*  __restrict__ out_f32,
                                    int coutr, int relu) {
  constexpr int WELEMS = 9 * 2 * NT * 32 * 16;   // packed weight elements
  extern __shared__ __bf16 smem[];
  __bf16* wlds  = smem;                          // [WELEMS]
  __bf16* actb0 = smem + WELEMS;                 // [ACTE]
  __bf16* actb1 = smem + WELEMS + ACTE;          // [ACTE]

  const int tid   = threadIdx.x;
  const int rowid = blockIdx.x;                  // b*H + h
  const int h     = rowid % Hh;

  // ---- stage packed weights into LDS (once per block, async) ----
  for (int i = tid; i < WELEMS / 8; i += 128)
    lds_fill16(&wlds[i * 8], wp + (size_t)i * 8);

  // ---- stage first act tile ----
  stage_act(actb0, in, rowid, h, 0, tid);
  wait_async_lds();
  __syncthreads();

  const int wave = tid >> 5;
  const int lane = tid & 31;
  const int m    = lane & 15;
  const int grp  = lane >> 4;

  for (int t = 0; t < TPW; ++t) {
    // prefetch next tile into the other buffer while we compute this one
    __bf16* cur = (t & 1) ? actb1 : actb0;
    __bf16* nxt = (t & 1) ? actb0 : actb1;
    if (t + 1 < TPW) stage_act(nxt, in, rowid, h, (t + 1) * TILE, tid);

    const int x0 = t * TILE;
    v8f acc[NT] = {};

    for (int dy = 0; dy < 3; ++dy) {
      for (int dx = 0; dx < 3; ++dx) {
#pragma unroll
        for (int kc = 0; kc < 2; ++kc) {
          // A fragment (16 pixels x K=32) from LDS: two aligned 16-B reads
          const int xl = wave * 16 + m + dx;
          const __bf16* ab = &cur[(size_t)((dy * 66 + xl) * Cch) + kc * 32 + grp * 8];
          v16bf a;
          ((uint4*)&a)[0] = *(const uint4*)(ab);        // K 0..7  (+grp*8)
          ((uint4*)&a)[1] = *(const uint4*)(ab + 16);   // K 16..23 (+grp*8)
          const int tap = dy * 3 + dx;
          // Batch-load all NT B fragments, then issue the WMMA burst,
          // so one DS wait covers NT matrix ops.
          v16bf bf[NT];
#pragma unroll
          for (int nt = 0; nt < NT; ++nt) {
            const __bf16* wb = wlds + (size_t)(((tap * 2 + kc) * NT + nt) * 32 + lane) * 16;
            ((uint4*)&bf[nt])[0] = *(const uint4*)(wb);
            ((uint4*)&bf[nt])[1] = *(const uint4*)(wb + 8);
          }
#pragma unroll
          for (int nt = 0; nt < NT; ++nt)
            acc[nt] = __builtin_amdgcn_wmma_f32_16x16x32_bf16(
                false, a, false, bf[nt], (short)0, acc[nt], false, false);
        }
      }
    }

    // ---- epilogue: BN, ReLU, residual, store (C/D: lane->N, VGPR r->M) ----
#pragma unroll
    for (int nt = 0; nt < NT; ++nt) {
      const int n = nt * 16 + m;
      float s = 1.f, o = 0.f;
      if (scale) { s = scale[n]; o = bias[n]; }
#pragma unroll
      for (int r = 0; r < 8; ++r) {
        const int pm = r + grp * 8;
        const int px = x0 + wave * 16 + pm;
        float v = acc[nt][r] * s + o;
        if (relu) v = fmaxf(v, 0.f);
        if (OUT_F32) {
          if (n < coutr)
            out_f32[((size_t)rowid * Ww + px) * coutr + n] = v;
        } else {
          size_t gi = ((size_t)rowid * Ww + px) * Cch + n;
          if (resid) v += (float)resid[gi];
          out_bf[gi] = (__bf16)v;
        }
      }
    }

    // next tile's async DMA must be complete, and all waves done with buffers
    wait_async_lds();
    __syncthreads();
  }
}

// ------------------------------------------------------------------
// adaptive filtering: vertical pass (softmax over taps 0..4)
// cl: [b][h][w][30] f32 ; base: NCHW f32 ; fh: NCHW f32
// ------------------------------------------------------------------
__global__ void filt_v_kernel(const float* __restrict__ cl, const float* __restrict__ base,
                              float* __restrict__ fh) {
  int idx = blockIdx.x * blockDim.x + threadIdx.x;
  if (idx >= Bn * 3 * Hh * Ww) return;
  int w  = idx % Ww;
  int h  = (idx / Ww) % Hh;
  int ch = (idx / (Ww * Hh)) % 3;
  int b  = idx / (Ww * Hh * 3);
  const float* wv = cl + ((size_t)(b * Hh + h) * Ww + w) * 30 + ch * 10;
  float mx = wv[0];
#pragma unroll
  for (int j = 1; j < 5; ++j) mx = fmaxf(mx, wv[j]);
  float e[5], sum = 0.f;
#pragma unroll
  for (int j = 0; j < 5; ++j) { e[j] = __expf(wv[j] - mx); sum += e[j]; }
  float inv = 1.f / sum, r = 0.f;
  const float* bp = base + (size_t)(b * 3 + ch) * Hh * Ww;
#pragma unroll
  for (int j = 0; j < 5; ++j) {
    int hs = h + j - 2;
    float bv = (hs >= 0 && hs < Hh) ? bp[(size_t)hs * Ww + w] : 0.f;
    r += bv * e[j] * inv;
  }
  fh[idx] = r;
}

// horizontal pass (softmax over taps 5..9) -> final output (NCHW f32)
__global__ void filt_h_kernel(const float* __restrict__ cl, const float* __restrict__ fh,
                              float* __restrict__ out) {
  int idx = blockIdx.x * blockDim.x + threadIdx.x;
  if (idx >= Bn * 3 * Hh * Ww) return;
  int w  = idx % Ww;
  int h  = (idx / Ww) % Hh;
  int ch = (idx / (Ww * Hh)) % 3;
  int b  = idx / (Ww * Hh * 3);
  const float* wv = cl + ((size_t)(b * Hh + h) * Ww + w) * 30 + ch * 10 + 5;
  float mx = wv[0];
#pragma unroll
  for (int j = 1; j < 5; ++j) mx = fmaxf(mx, wv[j]);
  float e[5], sum = 0.f;
#pragma unroll
  for (int j = 0; j < 5; ++j) { e[j] = __expf(wv[j] - mx); sum += e[j]; }
  float inv = 1.f / sum, r = 0.f;
  const float* fp = fh + ((size_t)(b * 3 + ch) * Hh + h) * Ww;
#pragma unroll
  for (int j = 0; j < 5; ++j) {
    int ws = w + j - 2;
    float fv = (ws >= 0 && ws < Ww) ? fp[ws] : 0.f;
    r += fv * e[j] * inv;
  }
  out[idx] = r;
}

// ------------------------------------------------------------------
extern "C" void kernel_launch(void* const* d_in, const int* in_sizes, int n_in,
                              void* d_out, int out_size, void* d_ws, size_t ws_size,
                              hipStream_t stream) {
  (void)in_sizes; (void)n_in; (void)out_size; (void)ws_size;
  const float* feature = (const float*)d_in[0];
  const float* base    = (const float*)d_in[1];
  const float* w_first = (const float*)d_in[2];
  const float* kb_w1   = (const float*)d_in[3];
  const float* kb_g1   = (const float*)d_in[4];
  const float* kb_b1   = (const float*)d_in[5];
  const float* kb_m1   = (const float*)d_in[6];
  const float* kb_v1   = (const float*)d_in[7];
  const float* kb_w2   = (const float*)d_in[8];
  const float* kb_g2   = (const float*)d_in[9];
  const float* kb_b2   = (const float*)d_in[10];
  const float* kb_m2   = (const float*)d_in[11];
  const float* kb_v2   = (const float*)d_in[12];
  const float* w_last  = (const float*)d_in[13];

  char* ws = (char*)d_ws;
  size_t off = 0;
  auto alloc = [&](size_t bytes) -> void* {
    void* p = ws + off;
    off += (bytes + 255) & ~(size_t)255;
    return p;
  };

  const size_t NPIX = (size_t)Bn * Hh * Ww;
  __bf16* yA   = (__bf16*)alloc(NPIX * Cch * sizeof(__bf16));
  __bf16* yB   = (__bf16*)alloc(NPIX * Cch * sizeof(__bf16));
  float*  cl   = (float*) alloc(NPIX * 30 * sizeof(float));
  float*  fh   = (float*) alloc((size_t)Bn * 3 * Hh * Ww * sizeof(float));
  const int PT  = 9 * 2 * 4 * 32 * 16;     // packed elems per 64->64 conv
  const int PTL = 9 * 2 * 2 * 32 * 16;     // packed elems for conv_last
  __bf16* pw1  = (__bf16*)alloc((size_t)Rres * PT * sizeof(__bf16));
  __bf16* pw2  = (__bf16*)alloc((size_t)Rres * PT * sizeof(__bf16));
  __bf16* pwl  = (__bf16*)alloc((size_t)PTL * sizeof(__bf16));
  float*  s1   = (float*)alloc(Rres * Cch * sizeof(float));
  float*  o1   = (float*)alloc(Rres * Cch * sizeof(float));
  float*  s2   = (float*)alloc(Rres * Cch * sizeof(float));
  float*  o2   = (float*)alloc(Rres * Cch * sizeof(float));

  // ---- prep ----
  bn_prep_kernel<<<1, 256, 0, stream>>>(kb_g1, kb_b1, kb_m1, kb_v1,
                                        kb_g2, kb_b2, kb_m2, kb_v2,
                                        s1, o1, s2, o2, Rres * Cch);
  for (int r = 0; r < Rres; ++r) {
    pack_w3_kernel<<<(PT + 255) / 256, 256, 0, stream>>>(
        kb_w1 + (size_t)r * Cch * Cch * 9, pw1 + (size_t)r * PT, 4, Cch);
    pack_w3_kernel<<<(PT + 255) / 256, 256, 0, stream>>>(
        kb_w2 + (size_t)r * Cch * Cch * 9, pw2 + (size_t)r * PT, 4, Cch);
  }
  pack_w3_kernel<<<(PTL + 255) / 256, 256, 0, stream>>>(w_last, pwl, 2, 30);

  // ---- conv_first: f32 NCHW -> bf16 NHWC, ReLU ----
  conv_first_kernel<<<(int)(NPIX / 32), 256, 0, stream>>>(feature, w_first, yA);

  // ---- residual WMMA blocks (one block per image row) ----
  const size_t shmem4 = (size_t)(9 * 2 * 4 * 32 * 16 + 2 * ACTE) * sizeof(__bf16); // 124416 B
  const size_t shmem2 = (size_t)(9 * 2 * 2 * 32 * 16 + 2 * ACTE) * sizeof(__bf16); //  87552 B
  dim3 cgrid(Bn * Hh);
  for (int r = 0; r < Rres; ++r) {
    conv3x3_wmma_kernel<4, false><<<cgrid, 128, shmem4, stream>>>(
        yA, pw1 + (size_t)r * PT, s1 + r * Cch, o1 + r * Cch,
        nullptr, yB, nullptr, Cch, 1);
    conv3x3_wmma_kernel<4, false><<<cgrid, 128, shmem4, stream>>>(
        yB, pw2 + (size_t)r * PT, s2 + r * Cch, o2 + r * Cch,
        yA, yA, nullptr, Cch, 1);
  }

  // ---- conv_last: 64 -> 30, f32 out ----
  conv3x3_wmma_kernel<2, true><<<cgrid, 128, shmem2, stream>>>(
      yA, pwl, nullptr, nullptr, nullptr, nullptr, cl, 30, 0);

  // ---- adaptive separable filtering ----
  int np = Bn * 3 * Hh * Ww;
  filt_v_kernel<<<(np + 255) / 256, 256, 0, stream>>>(cl, base, fh);
  filt_h_kernel<<<(np + 255) / 256, 256, 0, stream>>>(cl, fh, (float*)d_out);
}
